// GCN_23699629539721
// MI455X (gfx1250) — matrix operand order repaired and checked by reading, not compiled, not run on hardware
//
#include <hip/hip_runtime.h>
#include <hip/hip_bf16.h>

typedef __attribute__((ext_vector_type(16))) __bf16 v16bf;
typedef __attribute__((ext_vector_type(8)))  float  v8f;

__device__ __forceinline__ unsigned short bf_bits(float f) {
    // native fp32 -> bf16 (RNE) conversion instruction
    __bf16 b = (__bf16)f;
    return __builtin_bit_cast(unsigned short, b);
}

union BFrag {
    v16bf v;
    unsigned u[8];
};

// ---------------------------------------------------------------------------
// H[M x N] = X[M x K] @ W[K x N], bf16 inputs (converted on the fly), fp32 acc.
// One block = 256 threads = 8 waves; each wave owns a 16-row M stripe and all
// N/16 column tiles. W is staged transposed (bf16) in dynamic LDS.
// ---------------------------------------------------------------------------
template <int K, int N>
__global__ __launch_bounds__(256) void gemm_bf16_wmma(
    const float* __restrict__ X, const float* __restrict__ W,
    float* __restrict__ H, int M) {
    extern __shared__ unsigned short s_wt[];   // [N][K] transposed bf16

    const int tid = threadIdx.x;
    // Stage W transposed into LDS as bf16: s_wt[n*K + k] = bf16(W[k*N + n])
    for (int idx = tid; idx < N * K; idx += 256) {
        const int n = idx / K;
        const int k = idx - n * K;
        s_wt[idx] = bf_bits(W[(size_t)k * N + n]);
    }
    __syncthreads();

    const int wave = tid >> 5;
    const int lane = tid & 31;
    const int half = lane >> 4;       // 0: lanes 0-15, 1: lanes 16-31
    const int nsub = lane & 15;       // column within 16-wide tile / A row
    const int m0   = blockIdx.x * 128 + wave * 16;

    // A-fragment source row (lanes 0-15 and 16-31 both carry M = m0 + nsub)
    int arow = m0 + nsub;
    if (arow > M - 1) arow = M - 1;           // clamp: duplicate last row
    const float* __restrict__ xrow = X + (size_t)arow * K;

    constexpr int NT = N / 16;
    v8f acc[NT] = {};

    for (int kc = 0; kc < K; kc += 32) {
        // ---- A fragment: 16x32 bf16, contiguous k order (k = 16*half + s), same as B.
        // Lane needs K run [kc+half*16, +16): contiguous.
        const float4 r0 = *(const float4*)(xrow + kc + half * 16);
        const float4 r1 = *(const float4*)(xrow + kc + half * 16 + 4);
        const float4 r2 = *(const float4*)(xrow + kc + half * 16 + 8);
        const float4 r3 = *(const float4*)(xrow + kc + half * 16 + 12);
        v16bf av;
        av[0]  = (__bf16)r0.x; av[1]  = (__bf16)r0.y;
        av[2]  = (__bf16)r0.z; av[3]  = (__bf16)r0.w;
        av[4]  = (__bf16)r1.x; av[5]  = (__bf16)r1.y;
        av[6]  = (__bf16)r1.z; av[7]  = (__bf16)r1.w;
        av[8]  = (__bf16)r2.x; av[9]  = (__bf16)r2.y;
        av[10] = (__bf16)r2.z; av[11] = (__bf16)r2.w;
        av[12] = (__bf16)r3.x; av[13] = (__bf16)r3.y;
        av[14] = (__bf16)r3.z; av[15] = (__bf16)r3.w;

        // ---- B fragments from LDS: 32x16 bf16 per N-tile ----
#pragma unroll
        for (int nt = 0; nt < NT; ++nt) {
            BFrag b;
#pragma unroll
            for (int v = 0; v < 8; ++v) {
                const int k = kc + half * 16 + v * 2;
                b.u[v] = *(const unsigned*)&s_wt[(size_t)(nt * 16 + nsub) * K + k];
            }
            acc[nt] = __builtin_amdgcn_wmma_f32_16x16x32_bf16(
                false, av, false, b.v, (short)0, acc[nt], false, false);
        }
    }

    // ---- Store D: f32 16x16 layout (VGPR v -> row m0 + v + 8*half) ----
    if (m0 + 16 <= M) {
        // fast path: full tile, no guards
#pragma unroll
        for (int nt = 0; nt < NT; ++nt) {
#pragma unroll
            for (int v = 0; v < 8; ++v) {
                const int row = m0 + v + half * 8;
                H[(size_t)row * N + nt * 16 + nsub] = acc[nt][v];
            }
        }
    } else {
#pragma unroll
        for (int nt = 0; nt < NT; ++nt) {
#pragma unroll
            for (int v = 0; v < 8; ++v) {
                const int row = m0 + v + half * 8;
                if (row < M)
                    H[(size_t)row * N + nt * 16 + nsub] = acc[nt][v];
            }
        }
    }
}

// ---------------------------------------------------------------------------
// Edge kernels
// ---------------------------------------------------------------------------
__global__ __launch_bounds__(256) void zero_f32(float* __restrict__ p, long long n) {
    long long i = (long long)blockIdx.x * blockDim.x + threadIdx.x;
    const long long stride = (long long)gridDim.x * blockDim.x;
    for (; i < n; i += stride) p[i] = 0.0f;
}

__global__ __launch_bounds__(256) void degree_kernel(
    const int* __restrict__ edges, float* __restrict__ deg, int E) {
    const int e = blockIdx.x * blockDim.x + threadIdx.x;
    if (e < E) unsafeAtomicAdd(&deg[edges[E + e]], 1.0f);
}

__global__ __launch_bounds__(256) void dinv_kernel(float* __restrict__ deg, int Nn) {
    const int i = blockIdx.x * blockDim.x + threadIdx.x;
    if (i < Nn) deg[i] = rsqrtf(deg[i] + 1.0f);   // +1 self loop; always > 0
}

// One wave32 per edge; lane owns F/32 consecutive features.
template <int F>
__global__ __launch_bounds__(256) void edge_agg(
    const int* __restrict__ edges, const float* __restrict__ h,
    const float* __restrict__ dinv, float* __restrict__ agg, int E) {
    const int eidx = blockIdx.x * 8 + (threadIdx.x >> 5);
    if (eidx >= E) return;
    const int lane = threadIdx.x & 31;
    const int src = edges[eidx];
    const int dst = edges[E + eidx];
    const float norm = dinv[src] * dinv[dst];
    constexpr int PER = F / 32;                   // 4 (F=128) or 2 (F=64)
    const float* __restrict__ hs = h + (size_t)src * F + lane * PER;
    float* __restrict__ ad = agg + (size_t)dst * F + lane * PER;
    if (PER == 4) {
        const float4 hv = *(const float4*)hs;     // 16B-aligned gather
        unsafeAtomicAdd(&ad[0], norm * hv.x);
        unsafeAtomicAdd(&ad[1], norm * hv.y);
        unsafeAtomicAdd(&ad[2], norm * hv.z);
        unsafeAtomicAdd(&ad[3], norm * hv.w);
    } else {
        const float2 hv = *(const float2*)hs;
        unsafeAtomicAdd(&ad[0], norm * hv.x);
        unsafeAtomicAdd(&ad[1], norm * hv.y);
    }
}

// out = (agg + dinv^2 * h + bias), optional ReLU
template <bool RELU, int F>
__global__ __launch_bounds__(256) void finalize_kernel(
    const float* __restrict__ agg, const float* __restrict__ h,
    const float* __restrict__ dinv, const float* __restrict__ bias,
    float* __restrict__ out, int Nn) {
    const long long idx = (long long)blockIdx.x * blockDim.x + threadIdx.x;
    if (idx >= (long long)Nn * F) return;
    const int i = (int)(idx / F);
    const int f = (int)(idx - (long long)i * F);
    const float di = dinv[i];
    float v = agg[idx] + di * di * h[idx] + bias[f];
    if (RELU) v = fmaxf(v, 0.0f);
    out[idx] = v;
}

// ---------------------------------------------------------------------------
extern "C" void kernel_launch(void* const* d_in, const int* in_sizes, int n_in,
                              void* d_out, int out_size, void* d_ws, size_t ws_size,
                              hipStream_t stream) {
    const float* x     = (const float*)d_in[0];   // [N, 512]
    const int*   edges = (const int*)d_in[1];     // [2, E]
    const float* W1    = (const float*)d_in[2];   // [512, 128]
    const float* b1    = (const float*)d_in[3];   // [128]
    const float* W2    = (const float*)d_in[4];   // [128, 64]
    const float* b2    = (const float*)d_in[5];   // [64]

    const int Nn = in_sizes[0] / 512;
    const int E  = in_sizes[1] / 2;

    // Workspace layout (floats):
    //   dinv [Npad] | h1 [Nn*128] | agg1 [Nn*128]
    //   after finalize1: h1 region reused as h2 [Nn*64] + agg2 [Nn*64]
    float* ws   = (float*)d_ws;
    const long long Npad = ((long long)Nn + 255) / 256 * 256;
    float* dinv = ws;
    float* h1   = ws + Npad;
    float* agg1 = h1 + (long long)Nn * 128;
    float* h2   = h1;                          // reuse (h1 dead after finalize1)
    float* agg2 = h1 + (long long)Nn * 64;

    const dim3 blk(256);

    // degrees -> dinv
    zero_f32<<<1024, blk, 0, stream>>>(dinv, Nn);
    degree_kernel<<<(E + 255) / 256, blk, 0, stream>>>(edges, dinv, E);
    dinv_kernel<<<(Nn + 255) / 256, blk, 0, stream>>>(dinv, Nn);

    // ---- layer 1 ----
    zero_f32<<<4096, blk, 0, stream>>>(agg1, (long long)Nn * 128);
    gemm_bf16_wmma<512, 128><<<(Nn + 127) / 128, blk, 512 * 128 * 2, stream>>>(
        x, W1, h1, Nn);
    edge_agg<128><<<(E + 7) / 8, blk, 0, stream>>>(edges, h1, dinv, agg1, E);
    // relu(agg1 + dinv^2*h1 + b1) written in-place into agg1 (input to layer 2)
    finalize_kernel<true, 128><<<(int)(((long long)Nn * 128 + 255) / 256), blk, 0, stream>>>(
        agg1, h1, dinv, b1, agg1, Nn);

    // ---- layer 2 ----
    gemm_bf16_wmma<128, 64><<<(Nn + 127) / 128, blk, 128 * 64 * 2, stream>>>(
        agg1, W2, h2, Nn);
    zero_f32<<<4096, blk, 0, stream>>>(agg2, (long long)Nn * 64);
    edge_agg<64><<<(E + 7) / 8, blk, 0, stream>>>(edges, h2, dinv, agg2, E);
    finalize_kernel<false, 64><<<(int)(((long long)Nn * 64 + 255) / 256), blk, 0, stream>>>(
        agg2, h2, dinv, b2, (float*)d_out, Nn);
}
